// NCMHAtt_83468394430963
// MI455X (gfx1250) — matrix operand order repaired and checked by reading, not compiled, not run on hardware
//
#include <hip/hip_runtime.h>
#include <hip/hip_bf16.h>
#include <math.h>
#include <stdint.h>

// Problem constants (from reference): B=4, S=1024, H=1024, NH=16, DH=64, G=100
#define B_   4
#define S_   1024
#define H_   1024
#define NH_  16
#define DH_  64
#define G_   100

typedef __attribute__((ext_vector_type(16))) __bf16 v16bf;
typedef __attribute__((ext_vector_type(8)))  __bf16 v8bf;
typedef __attribute__((ext_vector_type(8)))  float  v8f;
typedef __attribute__((ext_vector_type(4)))  unsigned int u32x4;
typedef __attribute__((ext_vector_type(8)))  int i32x8;
typedef __attribute__((ext_vector_type(4)))  int i32x4;

#if defined(__gfx1250__) && __has_builtin(__builtin_amdgcn_tensor_load_to_lds) && \
    __has_builtin(__builtin_amdgcn_s_wait_tensorcnt)
#define USE_TDM 1
#else
#define USE_TDM 0
#endif

// ---------------------------------------------------------------------------
// CDNA5 wave32 WMMA: D = A(16x32 bf16) * B(32x16 bf16) + C(16x16 f32)
// Operand layouts used here (A and B share contiguous k order, not the ISA table order):
//   A: m = lane&15,  k(e) = e + ((lane>>4)<<4)
//      -> per lane: one contiguous 16-elem k-run at 16*half (two 8-elem loads)
//   B: n = lane&15,  k(e) = e + ((lane>>4)<<4)
//      -> per lane: one contiguous 16-elem k-run at 16*half
//   C/D: m = r + ((lane>>4)<<3), n = lane&15
// ---------------------------------------------------------------------------
__device__ __forceinline__ v8f wmma_bf16(v16bf a, v16bf b, v8f c) {
  return __builtin_amdgcn_wmma_f32_16x16x32_bf16(
      false, a, false, b, (short)0, c, false, false);
}

__device__ __forceinline__ v16bf cat8(v8bf lo, v8bf hi) {
  return __builtin_shufflevector(lo, hi, 0, 1, 2, 3, 4, 5, 6, 7,
                                 8, 9, 10, 11, 12, 13, 14, 15);
}

__device__ __forceinline__ unsigned lds_off(const void* p) {
  return (unsigned)(uintptr_t)p;  // LDS aperture: addr[31:0] is the LDS offset
}

#if USE_TDM
// Constant part of a 2D Tensor-DMA descriptor (bf16 elements).
__device__ __forceinline__ i32x8 tdm_g1_2d(unsigned tile_d0, unsigned tile_d1,
                                           unsigned tensor_d0, unsigned tensor_d1,
                                           unsigned stride0) {
  i32x8 g1;
  g1[0] = (1 << 16);                                    // data_size=1 (2 bytes)
  g1[1] = (int)((tensor_d0 & 0xFFFFu) << 16);           // tensor_dim0[15:0]
  g1[2] = (int)(((tensor_d0 >> 16) & 0xFFFFu) | ((tensor_d1 & 0xFFFFu) << 16));
  g1[3] = (int)(((tensor_d1 >> 16) & 0xFFFFu) | ((tile_d0 & 0xFFFFu) << 16));
  g1[4] = (int)(tile_d1 & 0xFFFFu);                     // tile_dim1 (tile_dim2=0)
  g1[5] = (int)stride0;                                 // tensor_dim0_stride[31:0]
  g1[6] = 0;
  g1[7] = 0;
  return g1;
}

// Issue TENSOR_LOAD_TO_LDS with per-call addresses + precomputed group 1.
// clang-23 builtin arity: (g0:u32x4, g1:i32x8, g2:i32x4, g3:i32x4, i32x8, cpol)
__device__ __forceinline__ void tdm_issue(unsigned lds, const void* g, i32x8 g1) {
  uint64_t ga = (uint64_t)(uintptr_t)g;
  u32x4 g0;
  g0.x = 1u;                                            // count=1, user mode
  g0.y = lds;                                           // lds_addr
  g0.z = (unsigned)ga;                                  // global_addr[31:0]
  g0.w = (unsigned)((ga >> 32) & 0x01FFFFFFu) | (2u << 30);  // addr[56:32] | type=2
  i32x4 z4 = {0, 0, 0, 0};
  i32x8 z8 = {0, 0, 0, 0, 0, 0, 0, 0};
  __builtin_amdgcn_tensor_load_to_lds(g0, g1, z4, z4, z8, 0);
}
#endif

// ---------------------------------------------------------------------------
// Kernel 0a: pre-transpose a weight matrix to bf16: Wt[n][k] = (bf16)W[k][n]
// ---------------------------------------------------------------------------
__global__ __launch_bounds__(256)
void transpose_w_bf16(const float* __restrict__ W, __bf16* __restrict__ Wt) {
  __shared__ float T[32][33];
  const int bx = blockIdx.x * 32;   // n tile
  const int by = blockIdx.y * 32;   // k tile
  const int tx = threadIdx.x & 31, ty = threadIdx.x >> 5;  // 32x8
#pragma unroll
  for (int i = 0; i < 32; i += 8)
    T[ty + i][tx] = W[(size_t)(by + ty + i) * H_ + bx + tx];  // T[k'][n']
  __syncthreads();
#pragma unroll
  for (int i = 0; i < 32; i += 8)
    Wt[(size_t)(bx + ty + i) * H_ + by + tx] = (__bf16)T[tx][ty + i];
}

// ---------------------------------------------------------------------------
// Kernel 0b: additive mask: maskAdd[b*S+key] = mask ? -1e9 : 0
// ---------------------------------------------------------------------------
__global__ __launch_bounds__(256)
void build_maskadd(const unsigned char* __restrict__ mask, float* __restrict__ madd) {
  int i = blockIdx.x * 256 + threadIdx.x;
  if (i < B_ * S_) madd[i] = mask[i] ? -1e9f : 0.0f;
}

// ---------------------------------------------------------------------------
// Kernel 1: Y = scaleOut * (X(f32)[4096,1024] @ W + bias) via Wt(bf16)[n][k].
// No LDS. Per wave: 32(M) x 64(N) tile -> each B-fragment feeds two WMMAs.
// Output bf16: transposeOut=0 -> [B][NH][S][DH]; =1 -> [B][NH][DH][S] (for V).
// Note: a 256-row tile never crosses a batch, a 64-col tile never crosses a
// head, so batch/head indices are block-uniform and element offsets are
// compile-time immediates off one per-lane base pointer.
// ---------------------------------------------------------------------------
__global__ __launch_bounds__(256)
void proj_to_heads(const float* __restrict__ X, const __bf16* __restrict__ Wt,
                   const float* __restrict__ bias, __bf16* __restrict__ outH,
                   int transposeOut, float scaleOut) {
  const int tid  = threadIdx.x;
  const int wave = tid >> 5, lane = tid & 31;
  const int half = lane >> 4, n15 = lane & 15;
  const int m0 = blockIdx.y * 256;           // 8 waves x 32 rows
  const int n0 = blockIdx.x * 64;

  const float* xr0 = X + (size_t)(m0 + wave * 32 + n15) * H_;
  const float* xr1 = xr0 + (size_t)16 * H_;
  v8f acc[2][4] = {};

  for (int k0 = 0; k0 < H_; k0 += 32) {
    __builtin_prefetch(xr0 + k0 + 64, 0, 1);
    __builtin_prefetch(xr1 + k0 + 64, 0, 1);
    v16bf a[2];
#pragma unroll
    for (int ms = 0; ms < 2; ++ms) {
      const float* xr = ms ? xr1 : xr0;
      v8f f0 = *(const v8f*)(xr + k0 + half * 16);
      v8f f1 = *(const v8f*)(xr + k0 + 8 + half * 16);
#pragma unroll
      for (int e = 0; e < 8; ++e) { a[ms][e] = (__bf16)f0[e]; a[ms][e + 8] = (__bf16)f1[e]; }
    }
#pragma unroll
    for (int nt = 0; nt < 4; ++nt) {
      v16bf b = *(const v16bf*)(Wt + (size_t)(n0 + nt * 16 + n15) * H_ + k0 + (half << 4));
#pragma unroll
      for (int ms = 0; ms < 2; ++ms)
        acc[ms][nt] = wmma_bf16(a[ms], b, acc[ms][nt]);
    }
  }

  // ---- epilogue: bias + scale, single uniform layout branch ----
  float bs[4];
#pragma unroll
  for (int nt = 0; nt < 4; ++nt) bs[nt] = bias[n0 + nt * 16 + n15];

  const int bb = m0 >> 10;                        // block-uniform batch
  const int hh = n0 >> 6;                         // block-uniform head
  const int s0 = (m0 & (S_ - 1)) + wave * 32 + (half << 3);

  if (transposeOut) {
    // [B][NH][DH][S]: elem (d = nt*16 + n15, s = s0 + ms*16 + r)
    __bf16* obase = outH + (((size_t)(bb * NH_ + hh) * DH_ + n15) << 10) + s0;
#pragma unroll
    for (int nt = 0; nt < 4; ++nt)
#pragma unroll
      for (int ms = 0; ms < 2; ++ms)
#pragma unroll
        for (int r = 0; r < 8; ++r)
          obase[((size_t)(nt * 16) << 10) + ms * 16 + r] =
              (__bf16)((acc[ms][nt][r] + bs[nt]) * scaleOut);
  } else {
    // [B][NH][S][DH]: elem (s = s0 + ms*16 + r, d = nt*16 + n15)
    __bf16* obase = outH + ((((size_t)(bb * NH_ + hh) << 10) + s0) << 6) + n15;
#pragma unroll
    for (int nt = 0; nt < 4; ++nt)
#pragma unroll
      for (int ms = 0; ms < 2; ++ms)
#pragma unroll
        for (int r = 0; r < 8; ++r)
          obase[(size_t)(ms * 16 + r) * DH_ + nt * 16] =
              (__bf16)((acc[ms][nt][r] + bs[nt]) * scaleOut);
  }
}

// ---------------------------------------------------------------------------
// Kernel 2: graph-modulated masked flash attention (Q pre-scaled by 1/sqrt(DH)).
// Q fragments live in registers; K ([key][d]) and V (pre-transposed [d][key])
// tiles stream through LDS via double-buffered TDM (b128 cooperative fallback).
// Row sums come from a WMMA against a ones-matrix; only row max uses shuffles.
// ---------------------------------------------------------------------------
__global__ __launch_bounds__(256)
void attn_kernel(const __bf16* __restrict__ qh, const __bf16* __restrict__ kh,
                 const __bf16* __restrict__ vhT, const float* __restrict__ maskAdd,
                 const float* __restrict__ graph, __bf16* __restrict__ attout) {
  __shared__ __bf16 Kbuf[2][32 * DH_];   // [key][d], packed
  __shared__ __bf16 Vbuf[2][DH_ * 32];   // [d][key], packed
  __shared__ __bf16 Ps[8][16][32];       // per-wave P transpose buffer

  const int tid  = threadIdx.x;
  const int wave = tid >> 5, lane = tid & 31;
  const int half = lane >> 4, n15 = lane & 15;
  const int bb = blockIdx.z, hh = blockIdx.y;
  const int q0 = blockIdx.x * 128;

  const __bf16* Qb = qh  + ((size_t)(bb * NH_ + hh) * S_) * DH_;
  const __bf16* Kb = kh  + ((size_t)(bb * NH_ + hh) * S_) * DH_;
  const __bf16* Vt = vhT + ((size_t)(bb * NH_ + hh) * DH_) * S_;

  // Q A-fragments direct from global (two aligned 16B runs per chunk)
  const __bf16* qrow = Qb + (size_t)(q0 + wave * 16 + n15) * DH_;
  v16bf qa[2];
#pragma unroll
  for (int cc = 0; cc < 2; ++cc)
    qa[cc] = cat8(*(const v8bf*)(qrow + cc * 32 + half * 16),
                  *(const v8bf*)(qrow + cc * 32 + 8 + half * 16));

  v16bf ones;
#pragma unroll
  for (int e = 0; e < 16; ++e) ones[e] = (__bf16)1.0f;

  float mrow[8], lrow[8];
  v8f o[4] = {};
#pragma unroll
  for (int r = 0; r < 8; ++r) { mrow[r] = -INFINITY; lrow[r] = 0.0f; }

#if USE_TDM
  const i32x8 g1K = tdm_g1_2d(DH_, 32, DH_, S_, DH_);   // K: [key][d] rows
  const i32x8 g1V = tdm_g1_2d(32, DH_, S_, DH_, S_);    // V: [d][key] rows
  int cur = 0;
  if (wave == 0) {
    tdm_issue(lds_off(&Kbuf[0][0]), Kb, g1K);
    tdm_issue(lds_off(&Vbuf[0][0]), Vt, g1V);
  }
#endif

  for (int kb = 0; kb < S_; kb += 32) {
#if USE_TDM
    if (wave == 0) __builtin_amdgcn_s_wait_tensorcnt(0);
    __syncthreads();                       // tile[cur] ready, tile[cur^1] free
    if (wave == 0 && kb + 32 < S_) {
      tdm_issue(lds_off(&Kbuf[cur ^ 1][0]), Kb + (size_t)(kb + 32) * DH_, g1K);
      tdm_issue(lds_off(&Vbuf[cur ^ 1][0]), Vt + (kb + 32), g1V);
    }
    const __bf16* Ktile = &Kbuf[cur][0];
    const __bf16* Vtile = &Vbuf[cur][0];
#else
    __syncthreads();
    {
      int o8 = tid * 8;                    // 2048 elems each, 8 per thread
      int kr = o8 >> 6, kc = o8 & 63;
      *(v8bf*)&Kbuf[0][o8] = *(const v8bf*)(Kb + (size_t)(kb + kr) * DH_ + kc);
      int vr = o8 >> 5, vc = o8 & 31;
      *(v8bf*)&Vbuf[0][o8] = *(const v8bf*)(Vt + (size_t)vr * S_ + kb + vc);
    }
    __syncthreads();
    const __bf16* Ktile = &Kbuf[0][0];
    const __bf16* Vtile = &Vbuf[0][0];
#endif

    // ---- scores: 16 queries x 32 keys (Q already carries 1/sqrt(DH)) ----
    float sc[2][8];
#pragma unroll
    for (int nblk = 0; nblk < 2; ++nblk) {
      v8f s = {};
#pragma unroll
      for (int cc = 0; cc < 2; ++cc) {
        v16bf bk = *(const v16bf*)(Ktile + (nblk * 16 + n15) * DH_ + cc * 32 + (half << 4));
        s = wmma_bf16(qa[cc], bk, s);
      }
#pragma unroll
      for (int r = 0; r < 8; ++r) sc[nblk][r] = s[r];
    }

    // ---- graph modulation: only the (q<100, key<100) corner blocks ----
    if (q0 < G_ && kb < G_) {
#pragma unroll
      for (int nblk = 0; nblk < 2; ++nblk) {
#pragma unroll
        for (int r = 0; r < 8; ++r) {
          int qq  = q0 + wave * 16 + r + (half << 3);
          int key = kb + nblk * 16 + n15;
          if (qq < G_ && key < G_)
            sc[nblk][r] *= (1.0f - graph[(size_t)(bb * G_ + qq) * G_ + key]);
        }
      }
    }

    // ---- additive mask (one value per key, loaded once per block) ----
    float madd0 = maskAdd[bb * S_ + kb + n15];
    float madd1 = maskAdd[bb * S_ + kb + 16 + n15];
#pragma unroll
    for (int r = 0; r < 8; ++r) { sc[0][r] += madd0; sc[1][r] += madd1; }

    // ---- online softmax: row max via half-wave shuffles ----
    float alpha[8];
#pragma unroll
    for (int r = 0; r < 8; ++r) {
      float bm = fmaxf(sc[0][r], sc[1][r]);
      for (int off = 1; off < 16; off <<= 1)
        bm = fmaxf(bm, __shfl_xor(bm, off, 32));
      float mnew = fmaxf(mrow[r], bm);
      alpha[r] = __expf(mrow[r] - mnew);
      sc[0][r] = __expf(sc[0][r] - mnew);
      sc[1][r] = __expf(sc[1][r] - mnew);
      mrow[r] = mnew;
#pragma unroll
      for (int ot = 0; ot < 4; ++ot) o[ot][r] *= alpha[r];
    }

    // ---- P: C-layout -> A-layout via per-wave LDS ----
#pragma unroll
    for (int nblk = 0; nblk < 2; ++nblk)
#pragma unroll
      for (int r = 0; r < 8; ++r)
        Ps[wave][r + (half << 3)][nblk * 16 + n15] = (__bf16)sc[nblk][r];
    asm volatile("s_wait_dscnt 0" ::: "memory");   // in-wave LDS RAW

    v16bf pa = cat8(*(const v8bf*)(&Ps[wave][n15][half * 16]),
                    *(const v8bf*)(&Ps[wave][n15][8 + half * 16]));

    // ---- row sums via WMMA against ones (replaces shuffle reduction) ----
    v8f zero = {};
    v8f rs = wmma_bf16(pa, ones, zero);
#pragma unroll
    for (int r = 0; r < 8; ++r) lrow[r] = lrow[r] * alpha[r] + rs[r];

    // ---- O += P @ V ----
#pragma unroll
    for (int ot = 0; ot < 4; ++ot) {
      v16bf bv = *(const v16bf*)(Vtile + (ot * 16 + n15) * 32 + (half << 4));
      o[ot] = wmma_bf16(pa, bv, o[ot]);
    }
#if USE_TDM
    cur ^= 1;
#endif
  }

  // ---- normalize and write [B][S][H] bf16 (h = head*64 + d) ----
  float rinv[8];
#pragma unroll
  for (int r = 0; r < 8; ++r) rinv[r] = 1.0f / lrow[r];
  __bf16* obase = attout + (((size_t)(bb << 10) + q0 + wave * 16 + (half << 3)) << 10)
                + hh * DH_ + n15;
#pragma unroll
  for (int ot = 0; ot < 4; ++ot)
#pragma unroll
    for (int r = 0; r < 8; ++r)
      obase[((size_t)r << 10) + ot * 16] = (__bf16)(o[ot][r] * rinv[r]);
}

// ---------------------------------------------------------------------------
// Kernel 3: out(f32)[4096,1024] = X(bf16)[4096,1024] @ Wm + bm  (via WmT bf16)
// Per wave: 32(M) x 64(N) tile.
// ---------------------------------------------------------------------------
__global__ __launch_bounds__(256)
void proj_out(const __bf16* __restrict__ X, const __bf16* __restrict__ Wt,
              const float* __restrict__ bias, float* __restrict__ out) {
  const int tid  = threadIdx.x;
  const int wave = tid >> 5, lane = tid & 31;
  const int half = lane >> 4, n15 = lane & 15;
  const int m0 = blockIdx.y * 256;
  const int n0 = blockIdx.x * 64;

  const __bf16* xr0 = X + (size_t)(m0 + wave * 32 + n15) * H_;
  const __bf16* xr1 = xr0 + (size_t)16 * H_;
  v8f acc[2][4] = {};

  for (int k0 = 0; k0 < H_; k0 += 32) {
    __builtin_prefetch(xr0 + k0 + 64, 0, 1);
    __builtin_prefetch(xr1 + k0 + 64, 0, 1);
    v16bf a[2];
#pragma unroll
    for (int ms = 0; ms < 2; ++ms) {
      const __bf16* xr = ms ? xr1 : xr0;
      a[ms] = cat8(*(const v8bf*)(xr + k0 + half * 16),
                   *(const v8bf*)(xr + k0 + 8 + half * 16));
    }
#pragma unroll
    for (int nt = 0; nt < 4; ++nt) {
      v16bf b = *(const v16bf*)(Wt + (size_t)(n0 + nt * 16 + n15) * H_ + k0 + (half << 4));
#pragma unroll
      for (int ms = 0; ms < 2; ++ms)
        acc[ms][nt] = wmma_bf16(a[ms], b, acc[ms][nt]);
    }
  }

  float bs[4];
#pragma unroll
  for (int nt = 0; nt < 4; ++nt) bs[nt] = bias[n0 + nt * 16 + n15];
  float* obase = out + (size_t)(m0 + wave * 32 + (half << 3)) * H_ + n0 + n15;
#pragma unroll
  for (int nt = 0; nt < 4; ++nt)
#pragma unroll
    for (int ms = 0; ms < 2; ++ms)
#pragma unroll
      for (int r = 0; r < 8; ++r)
        obase[(size_t)(ms * 16 + r) * H_ + nt * 16] = acc[ms][nt][r] + bs[nt];
}

// ---------------------------------------------------------------------------
// Host-side launcher
// d_in order: v, k, q, mask, graph, Wq, bq, Wk, bk, Wv, bv, Wm, bm
// Workspace: qh/kh/vhT/att_out (8MB bf16 each) + 4 transposed weights (2MB
// bf16 each) + maskAdd (16KB f32)
// ---------------------------------------------------------------------------
extern "C" void kernel_launch(void* const* d_in, const int* in_sizes, int n_in,
                              void* d_out, int out_size, void* d_ws, size_t ws_size,
                              hipStream_t stream) {
  (void)in_sizes; (void)n_in; (void)out_size; (void)ws_size;
  const float* v  = (const float*)d_in[0];
  const float* k  = (const float*)d_in[1];
  const float* q  = (const float*)d_in[2];
  const unsigned char* mask = (const unsigned char*)d_in[3];  // jax bool_: 1 byte
  const float* graph = (const float*)d_in[4];
  const float* Wq = (const float*)d_in[5];
  const float* bq = (const float*)d_in[6];
  const float* Wk = (const float*)d_in[7];
  const float* bk = (const float*)d_in[8];
  const float* Wv = (const float*)d_in[9];
  const float* bv = (const float*)d_in[10];
  const float* Wm = (const float*)d_in[11];
  const float* bm = (const float*)d_in[12];
  float* out = (float*)d_out;

  const size_t HEAD_ELEMS = (size_t)B_ * NH_ * S_ * DH_;  // 4,194,304
  const size_t W_ELEMS    = (size_t)H_ * H_;              // 1,048,576
  __bf16* qh  = (__bf16*)d_ws;
  __bf16* kh  = qh  + HEAD_ELEMS;
  __bf16* vhT = kh  + HEAD_ELEMS;
  __bf16* ao  = vhT + HEAD_ELEMS;
  __bf16* wtq = ao  + HEAD_ELEMS;
  __bf16* wtk = wtq + W_ELEMS;
  __bf16* wtv = wtk + W_ELEMS;
  __bf16* wtm = wtv + W_ELEMS;
  float*  madd = (float*)(wtm + W_ELEMS);

  dim3 gtr(H_ / 32, H_ / 32, 1);             // (32, 32)
  transpose_w_bf16<<<gtr, 256, 0, stream>>>(Wq, wtq);
  transpose_w_bf16<<<gtr, 256, 0, stream>>>(Wk, wtk);
  transpose_w_bf16<<<gtr, 256, 0, stream>>>(Wv, wtv);
  transpose_w_bf16<<<gtr, 256, 0, stream>>>(Wm, wtm);
  build_maskadd<<<(B_ * S_ + 255) / 256, 256, 0, stream>>>(mask, madd);

  dim3 gproj(H_ / 64, (B_ * S_) / 256, 1);   // (16, 16)
  proj_to_heads<<<gproj, 256, 0, stream>>>(q, wtq, bq, qh, 0, 0.125f);  // 1/sqrt(DH)
  proj_to_heads<<<gproj, 256, 0, stream>>>(k, wtk, bk, kh, 0, 1.0f);
  proj_to_heads<<<gproj, 256, 0, stream>>>(v, wtv, bv, vhT, 1, 1.0f);

  dim3 gattn(S_ / 128, NH_, B_);             // (8, 16, 4)
  attn_kernel<<<gattn, 256, 0, stream>>>(qh, kh, vhT, madd, graph, ao);

  proj_out<<<gproj, 256, 0, stream>>>(ao, wtm, bm, out);
}
